// MSTGCN_Block_21182778704673
// MI455X (gfx1250) — matrix operand order repaired and checked by reading, not compiled, not run on hardware
//
#include <hip/hip_runtime.h>

// MSTGCN block for MI455X (gfx1250, wave32).
// Problem sizes (fixed by the reference):
#define Bn  16
#define Tn  265
#define Vn  62
#define Fn  5
#define Kc  3
#define FCn 64
#define FTn 64
#define KTn 3

typedef __attribute__((ext_vector_type(16))) _Float16 v16h;
typedef __attribute__((ext_vector_type(8)))  float    v8f;

__device__ __forceinline__ float sigmoidf_(float x) { return 1.0f / (1.0f + __expf(-x)); }

// ---------------------------------------------------------------------------
// Temporal attention pieces
// ---------------------------------------------------------------------------

// t_lhs[b,t,:] = (sum_v x[b,t,v,f]*U1[v]) @ U2   -> (B,T,V)
__global__ __launch_bounds__(64) void k_tlhs(const float* __restrict__ x,
                                             const float* __restrict__ U1,
                                             const float* __restrict__ U2,
                                             float* __restrict__ tlhs) {
  __shared__ float xs[Vn * Fn];
  __shared__ float tf[Fn];
  int bid = blockIdx.x, tid = threadIdx.x;               // bid = b*Tn + t
  const float* xbt = x + (size_t)bid * Vn * Fn;
  for (int e = tid; e < Vn * Fn; e += 64) xs[e] = xbt[e];
  __syncthreads();
  if (tid < Fn) {
    float s = 0.f;
    for (int v = 0; v < Vn; ++v) s += xs[v * Fn + tid] * U1[v];
    tf[tid] = s;
  }
  __syncthreads();
  for (int j = tid; j < Vn; j += 64) {
    float s = 0.f;
    for (int f = 0; f < Fn; ++f) s += tf[f] * U2[f * Vn + j];
    tlhs[(size_t)bid * Vn + j] = s;
  }
}

// t_rhs[b,v,t] = sum_f x[b,t,v,f]*U3[f]   -> (B,V,T)
__global__ void k_trhs(const float* __restrict__ x, const float* __restrict__ U3,
                       float* __restrict__ trhs) {
  int idx = blockIdx.x * 256 + threadIdx.x;
  if (idx >= Bn * Tn * Vn) return;
  int b = idx / (Tn * Vn), rem = idx % (Tn * Vn);
  int t = rem / Vn, v = rem % Vn;
  float s = 0.f;
  for (int f = 0; f < Fn; ++f) s += x[(size_t)idx * Fn + f] * U3[f];
  trhs[((size_t)b * Vn + v) * Tn + t] = s;
}

// prodT[b,i,k] = t_lhs[b,i,:] . t_rhs[b,:,k]    (B,T,T)
__global__ __launch_bounds__(128) void k_prodT(const float* __restrict__ tlhs,
                                               const float* __restrict__ trhs,
                                               float* __restrict__ prodT) {
  __shared__ float lhs[Vn];
  int bid = blockIdx.x, b = bid / Tn, tid = threadIdx.x;
  for (int e = tid; e < Vn; e += 128) lhs[e] = tlhs[(size_t)bid * Vn + e];
  __syncthreads();
  const float* rb = trhs + (size_t)b * Vn * Tn;
  for (int k2 = tid; k2 < Tn; k2 += 128) {
    float s = 0.f;
    for (int j = 0; j < Vn; ++j) s += lhs[j] * rb[(size_t)j * Tn + k2];
    prodT[(size_t)bid * Tn + k2] = s;
  }
}

// Sig[b,i,k] = sigmoid( V_e[i,:] . prodT[b,:,k] + b_e[i,k] )
__global__ __launch_bounds__(128) void k_sigT(const float* __restrict__ prodT,
                                              const float* __restrict__ V_e,
                                              const float* __restrict__ b_e,
                                              float* __restrict__ Sig) {
  __shared__ float ve[Tn];
  int bid = blockIdx.x, b = bid / Tn, i = bid % Tn, tid = threadIdx.x;
  for (int e = tid; e < Tn; e += 128) ve[e] = V_e[(size_t)i * Tn + e];
  __syncthreads();
  const float* pb = prodT + (size_t)b * Tn * Tn;
  for (int k2 = tid; k2 < Tn; k2 += 128) {
    float s = 0.f;
    for (int j = 0; j < Tn; ++j) s += ve[j] * pb[(size_t)j * Tn + k2];
    s += b_e[(size_t)i * Tn + k2];
    Sig[(size_t)bid * Tn + k2] = sigmoidf_(s);
  }
}

// In-place column softmax over axis 1: S (B,N,M), softmax over i per (b,m).
__global__ __launch_bounds__(128) void k_softcol(float* __restrict__ S, int N, int M) {
  __shared__ float red[128];
  int blk = blockIdx.x, b = blk / M, col = blk % M, tid = threadIdx.x;
  float* base = S + (size_t)b * N * M + col;
  float m = -1e30f;
  for (int i = tid; i < N; i += 128) m = fmaxf(m, base[(size_t)i * M]);
  red[tid] = m; __syncthreads();
  for (int s = 64; s > 0; s >>= 1) { if (tid < s) red[tid] = fmaxf(red[tid], red[tid + s]); __syncthreads(); }
  float mx = red[0]; __syncthreads();
  float sm = 0.f;
  for (int i = tid; i < N; i += 128) sm += __expf(base[(size_t)i * M] - mx);
  red[tid] = sm; __syncthreads();
  for (int s = 64; s > 0; s >>= 1) { if (tid < s) red[tid] += red[tid + s]; __syncthreads(); }
  float inv = 1.0f / red[0]; __syncthreads();
  for (int i = tid; i < N; i += 128) base[(size_t)i * M] = __expf(base[(size_t)i * M] - mx) * inv;
}

// y[b, i, t'] = sum_tau x_flat[b,i,tau]*TAtt[b,tau,t'] stored (B, V*F, T);
// readers reinterpret the same flat buffer as (B,T,V,F) = torch .view semantics.
__global__ __launch_bounds__(128) void k_xTAtt(const float* __restrict__ x,
                                               const float* __restrict__ TAtt,
                                               float* __restrict__ xT) {
  __shared__ float xr[Tn];
  int bid = blockIdx.x, b = bid / (Vn * Fn), i = bid % (Vn * Fn), tid = threadIdx.x;
  for (int e = tid; e < Tn; e += 128) xr[e] = x[((size_t)b * Tn + e) * (Vn * Fn) + i];
  __syncthreads();
  const float* tb = TAtt + (size_t)b * Tn * Tn;
  for (int tp = tid; tp < Tn; tp += 128) {
    float s = 0.f;
    for (int tau = 0; tau < Tn; ++tau) s += xr[tau] * tb[(size_t)tau * Tn + tp];
    xT[((size_t)b * (Vn * Fn) + i) * Tn + tp] = s;
  }
}

// ---------------------------------------------------------------------------
// Spatial attention pieces (xT read with (B,T,V,F) strides)
// ---------------------------------------------------------------------------

__global__ __launch_bounds__(64) void k_slhs(const float* __restrict__ xT,
                                             const float* __restrict__ W1,
                                             const float* __restrict__ W2,
                                             float* __restrict__ slhs) {
  __shared__ float red[64];
  __shared__ float tf[Fn];
  int bid = blockIdx.x, b = bid / Vn, v = bid % Vn, tid = threadIdx.x;
  float loc[Fn] = {0.f, 0.f, 0.f, 0.f, 0.f};
  for (int t = tid; t < Tn; t += 64) {
    float w = W1[t];
    const float* p = xT + ((size_t)b * Tn + t) * (Vn * Fn) + v * Fn;
    for (int f = 0; f < Fn; ++f) loc[f] += p[f] * w;
  }
  for (int f = 0; f < Fn; ++f) {
    red[tid] = loc[f]; __syncthreads();
    for (int s = 32; s > 0; s >>= 1) { if (tid < s) red[tid] += red[tid + s]; __syncthreads(); }
    if (tid == 0) tf[f] = red[0];
    __syncthreads();
  }
  for (int t2 = tid; t2 < Tn; t2 += 64) {
    float s = 0.f;
    for (int f = 0; f < Fn; ++f) s += tf[f] * W2[f * Tn + t2];
    slhs[((size_t)b * Vn + v) * Tn + t2] = s;
  }
}

__global__ void k_srhs(const float* __restrict__ xT, const float* __restrict__ W3,
                       float* __restrict__ srhs) {
  int idx = blockIdx.x * 256 + threadIdx.x;
  if (idx >= Bn * Tn * Vn) return;
  const float* p = xT + (size_t)idx * Fn;
  float s = 0.f;
  for (int f = 0; f < Fn; ++f) s += p[f] * W3[f];
  srhs[idx] = s;
}

__global__ __launch_bounds__(64) void k_prodS(const float* __restrict__ slhs,
                                              const float* __restrict__ srhs,
                                              float* __restrict__ prodS) {
  __shared__ float lhs[Tn];
  int bid = blockIdx.x, b = bid / Vn, tid = threadIdx.x;
  for (int e = tid; e < Tn; e += 64) lhs[e] = slhs[(size_t)bid * Tn + e];
  __syncthreads();
  const float* rb = srhs + (size_t)b * Tn * Vn;
  for (int j = tid; j < Vn; j += 64) {
    float s = 0.f;
    for (int t = 0; t < Tn; ++t) s += lhs[t] * rb[(size_t)t * Vn + j];
    prodS[(size_t)bid * Vn + j] = s;
  }
}

__global__ __launch_bounds__(64) void k_sigS(const float* __restrict__ prodS,
                                             const float* __restrict__ V_s,
                                             const float* __restrict__ b_s,
                                             float* __restrict__ SAtt) {
  int bid = blockIdx.x, b = bid / Vn, i = bid % Vn, tid = threadIdx.x;
  for (int j = tid; j < Vn; j += 64) {
    float s = 0.f;
    for (int m = 0; m < Vn; ++m) s += V_s[i * Vn + m] * prodS[((size_t)b * Vn + m) * Vn + j];
    s += b_s[i * Vn + j];
    SAtt[(size_t)bid * Vn + j] = sigmoidf_(s);
  }
}

// ---------------------------------------------------------------------------
// Fused graph-learning + dynamic & static Chebyshev conv per (b,t).
// Writes relu'd results in f16 (A-panel source for the WMMA tconv).
// ---------------------------------------------------------------------------
__global__ __launch_bounds__(256) void k_cheb(const float* __restrict__ x,
                                              const float* __restrict__ a_gl,
                                              const float* __restrict__ SAtt,
                                              const float* __restrict__ cheb,
                                              const float* __restrict__ ThGL,
                                              const float* __restrict__ ThSD,
                                              _Float16* __restrict__ outGL,
                                              _Float16* __restrict__ outSD) {
  __shared__ float xs[Vn * Fn];
  __shared__ float SA[Vn * Vn];
  __shared__ float Lm[Vn * Vn];   // Sg -> A -> L (in place)
  __shared__ float P2[Vn * Vn];
  __shared__ float rhs[Vn * Fn];
  __shared__ float outb[Vn * FCn];
  __shared__ float dsum[Vn];
  __shared__ float rowsum[Vn];
  __shared__ float agl[Fn];

  int bid = blockIdx.x, b = bid / Tn, tid = threadIdx.x;
  const float* xbt = x + (size_t)bid * (Vn * Fn);
  for (int e = tid; e < Vn * Fn; e += 256) xs[e] = xbt[e];
  if (tid < Fn) agl[tid] = a_gl[tid];
  for (int e = tid; e < Vn * Vn; e += 256) SA[e] = SAtt[(size_t)b * Vn * Vn + e];
  __syncthreads();

  // Sg[u][v] = exp(sum_f |x_u - x_v| * a)
  for (int e = tid; e < Vn * Vn; e += 256) {
    int u = e / Vn, v = e % Vn;
    float s = 0.f;
    for (int f = 0; f < Fn; ++f) s += fabsf(xs[u * Fn + f] - xs[v * Fn + f]) * agl[f];
    Lm[e] = __expf(s);
  }
  __syncthreads();
  for (int u = tid; u < Vn; u += 256) {
    float r = 0.f;
    for (int v = 0; v < Vn; ++v) r += Lm[u * Vn + v];
    rowsum[u] = r;
  }
  __syncthreads();
  for (int e = tid; e < Vn * Vn; e += 256) Lm[e] /= rowsum[e / Vn];
  __syncthreads();
  // A = min(Sg, Sg^T) in place (each unordered pair once)
  for (int e = tid; e < Vn * Vn; e += 256) {
    int u = e / Vn, v = e % Vn;
    if (u < v) {
      float m = fminf(Lm[u * Vn + v], Lm[v * Vn + u]);
      Lm[u * Vn + v] = m; Lm[v * Vn + u] = m;
    }
  }
  __syncthreads();
  for (int i = tid; i < Vn; i += 256) {
    float s = 0.f;
    for (int u = 0; u < Vn; ++u) s += Lm[u * Vn + i];
    dsum[i] = s;
  }
  __syncthreads();
  // L[i][j] = d[i]*dij - A[i][j] - dij  (in place)
  for (int e = tid; e < Vn * Vn; e += 256) {
    int i = e / Vn, j = e % Vn;
    float a = Lm[e];
    Lm[e] = (i == j) ? (dsum[i] - 1.0f - a) : (-a);
  }
  __syncthreads();
  // P2 = 2 L@L - I
  for (int e = tid; e < Vn * Vn; e += 256) {
    int i = e / Vn, j = e % Vn;
    float s = 0.f;
    for (int m = 0; m < Vn; ++m) s += Lm[i * Vn + m] * Lm[m * Vn + j];
    P2[e] = 2.0f * s - ((i == j) ? 1.0f : 0.0f);
  }
  __syncthreads();

  // ---- dynamic (GL) pass ----
  for (int e = tid; e < Vn * FCn; e += 256) outb[e] = 0.f;
  __syncthreads();
  for (int kk = 0; kk < Kc; ++kk) {
    for (int e = tid; e < Vn * Fn; e += 256) {
      int u = e / Fn, f = e % Fn;
      float s;
      if (kk == 0) {
        s = SA[u * Vn + u] * xs[u * Fn + f];
      } else {
        const float* P = (kk == 1) ? Lm : P2;
        s = 0.f;
        for (int v = 0; v < Vn; ++v) s += P[v * Vn + u] * SA[v * Vn + u] * xs[v * Fn + f];
      }
      rhs[e] = s;
    }
    __syncthreads();
    for (int e = tid; e < Vn * FCn; e += 256) {
      int u = e / FCn, o2 = e % FCn;
      float s = 0.f;
      for (int f = 0; f < Fn; ++f) s += rhs[u * Fn + f] * ThGL[(kk * Fn + f) * FCn + o2];
      outb[e] += s;
    }
    __syncthreads();
  }
  for (int e = tid; e < Vn * FCn; e += 256)
    outGL[(size_t)bid * Vn * FCn + e] = (_Float16)fmaxf(outb[e], 0.0f);
  __syncthreads();

  // ---- static (SD) pass ----
  for (int e = tid; e < Vn * FCn; e += 256) outb[e] = 0.f;
  __syncthreads();
  for (int kk = 0; kk < Kc; ++kk) {
    const float* C = cheb + (size_t)kk * Vn * Vn;
    for (int e = tid; e < Vn * Fn; e += 256) {
      int u = e / Fn, f = e % Fn;
      float s = 0.f;
      for (int v = 0; v < Vn; ++v) s += C[v * Vn + u] * SA[v * Vn + u] * xs[v * Fn + f];
      rhs[e] = s;
    }
    __syncthreads();
    for (int e = tid; e < Vn * FCn; e += 256) {
      int u = e / FCn, o2 = e % FCn;
      float s = 0.f;
      for (int f = 0; f < Fn; ++f) s += rhs[u * Fn + f] * ThSD[(kk * Fn + f) * FCn + o2];
      outb[e] += s;
    }
    __syncthreads();
  }
  for (int e = tid; e < Vn * FCn; e += 256)
    outSD[(size_t)bid * Vn * FCn + e] = (_Float16)fmaxf(outb[e], 0.0f);
}

// ---------------------------------------------------------------------------
// Repack conv weights (FT,FC,KT,1) -> f16 B FRAGMENTS in WMMA lane order:
// Bfrag[((ks*4+nt)*32 + lane)*16 + h], so each lane's 16 halves of a
// 32x16 B fragment are 32 contiguous bytes -> two global_load_b128.
// Lane layout: lanes 0-15 hold K = ks*32 + {0..7,16..23}, lanes 16-31 K = +{8..15,24..31},
// column N = nt*16 + (lane&15).
// ---------------------------------------------------------------------------
__global__ void k_prepW(const float* __restrict__ WcGL, const float* __restrict__ WcSD,
                        _Float16* __restrict__ BfGL, _Float16* __restrict__ BfSD) {
  int tid = blockIdx.x * blockDim.x + threadIdx.x;
  const int total = 6 * 4 * 32 * 16;          // ks * nt * lane * halves = 12288
  for (int e = tid; e < total; e += blockDim.x * gridDim.x) {
    int h = e & 15;
    int lane = (e >> 4) & 31;
    int fid = e >> 9;                         // 0..23
    int ks = fid >> 2, nt = fid & 3;
    int k = ks * 32 + ((lane & 16) ? 8 : 0) + (h & 7) + ((h & 8) ? 16 : 0);   // K index 0..191
    int n = nt * 16 + (lane & 15);                  // FT index 0..63
    int kt = k / FCn, fc = k % FCn;
    BfGL[e] = (_Float16)WcGL[(n * FCn + fc) * KTn + kt];
    BfSD[e] = (_Float16)WcSD[(n * FCn + fc) * KTn + kt];
  }
}

// ---------------------------------------------------------------------------
// WMMA temporal conv + bias + LayerNorm, fused. One wave per 16-row tile.
// GEMM: M = B*T*V, K = 192 (kt,fc), N = 64 (ft). 24 v_wmma per wave.
// A panel staged in LDS already in fragment lane order so each lane reads
// 32 contiguous bytes per k-step (2x ds_load_b128).
// ---------------------------------------------------------------------------
__global__ __launch_bounds__(32) void k_convln(const _Float16* __restrict__ src,
                                               const _Float16* __restrict__ Bf,
                                               const float* __restrict__ bias,
                                               const float* __restrict__ gamma,
                                               const float* __restrict__ beta,
                                               float* __restrict__ dst) {
  __shared__ alignas(32) _Float16 As[6 * 32 * 16];   // [ks][lane][h], 6 KB
  __shared__ float Cs[16 * 64];
  int tile = blockIdx.x, lane = threadIdx.x;

  // Stage A panel (16 rows x 192 K) in WMMA fragment order, zero-padded at
  // the time boundary.  Inverse of the 16-bit A layout:
  //   kl in {0..7,16..23} -> lanes 0-15 (group 0), {8..15,24..31} -> lanes 16-31
  //   h = (kl&7) + (kl>=16 ? 8 : 0)
  for (int e = lane; e < 16 * 192; e += 32) {
    int m = e / 192, k = e % 192;
    int kt = k / FCn, fc = k % FCn;
    int r = tile * 16 + m;
    int b = r / (Tn * Vn), rem = r % (Tn * Vn);
    int t = rem / Vn, v = rem % Vn;
    int tt = t + kt - 1;
    _Float16 val = (_Float16)0.0f;
    if (tt >= 0 && tt < Tn) val = src[((size_t)(b * Tn + tt) * Vn + v) * FCn + fc];
    int ks = k >> 5, kl = k & 31;
    int grp = (kl >> 3) & 1;
    int h = (kl & 7) + ((kl & 16) >> 1);
    int dlane = grp * 16 + m;
    As[(ks * 32 + dlane) * 16 + h] = val;
  }
  __syncthreads();

  v8f acc[4] = {};
  const v16h* Afr = (const v16h*)As;
  const v16h* Bfr = (const v16h*)Bf;
#pragma unroll
  for (int ks = 0; ks < 6; ++ks) {
    v16h a = Afr[ks * 32 + lane];
#pragma unroll
    for (int nt = 0; nt < 4; ++nt) {
      v16h bf = Bfr[(ks * 4 + nt) * 32 + lane];
      acc[nt] = __builtin_amdgcn_wmma_f32_16x16x32_f16(
          false, a, false, bf, (short)0, acc[nt], false, false);
    }
  }

  // Spill C tile (+bias) to LDS: VGPR i holds M=i (lanes 0-15) / M=i+8 (lanes 16-31).
  int ncol = lane & 15;
#pragma unroll
  for (int nt = 0; nt < 4; ++nt)
#pragma unroll
    for (int i = 0; i < 8; ++i) {
      int mm = (lane < 16) ? i : (i + 8);
      int nn = nt * 16 + ncol;
      Cs[mm * 64 + nn] = acc[nt][i] + bias[nn];
    }
  __syncthreads();

  // LayerNorm over FT per row, scatter to (B,FT,T,V).
  if (lane < 16) {
    int m = lane, r = tile * 16 + m;
    int b = r / (Tn * Vn), rem = r % (Tn * Vn);
    int t = rem / Vn, v = rem % Vn;
    float mu = 0.f;
    for (int o2 = 0; o2 < 64; ++o2) mu += Cs[m * 64 + o2];
    mu *= (1.0f / 64.0f);
    float var = 0.f;
    for (int o2 = 0; o2 < 64; ++o2) { float d = Cs[m * 64 + o2] - mu; var += d * d; }
    var *= (1.0f / 64.0f);
    float rstd = rsqrtf(var + 1e-5f);
    for (int o2 = 0; o2 < 64; ++o2)
      dst[((size_t)(b * FTn + o2) * Tn + t) * Vn + v] =
          (Cs[m * 64 + o2] - mu) * rstd * gamma[o2] + beta[o2];
  }
}

// ---------------------------------------------------------------------------

extern "C" void kernel_launch(void* const* d_in, const int* in_sizes, int n_in,
                              void* d_out, int out_size, void* d_ws, size_t ws_size,
                              hipStream_t stream) {
  (void)in_sizes; (void)n_in; (void)out_size; (void)ws_size;
  const float* x    = (const float*)d_in[0];
  const float* U1   = (const float*)d_in[1];
  const float* U2   = (const float*)d_in[2];
  const float* U3   = (const float*)d_in[3];
  const float* b_e  = (const float*)d_in[4];
  const float* V_e  = (const float*)d_in[5];
  const float* W1   = (const float*)d_in[6];
  const float* W2   = (const float*)d_in[7];
  const float* W3   = (const float*)d_in[8];
  const float* b_s  = (const float*)d_in[9];
  const float* V_s  = (const float*)d_in[10];
  const float* a_gl = (const float*)d_in[11];
  const float* ThGL = (const float*)d_in[12];
  const float* ThSD = (const float*)d_in[13];
  const float* cheb = (const float*)d_in[14];
  const float* WcGL = (const float*)d_in[15];
  const float* bcGL = (const float*)d_in[16];
  const float* WcSD = (const float*)d_in[17];
  const float* bcSD = (const float*)d_in[18];
  const float* gGL  = (const float*)d_in[19];
  const float* beGL = (const float*)d_in[20];
  const float* gSD  = (const float*)d_in[21];
  const float* beSD = (const float*)d_in[22];

  float* ws = (float*)d_ws;
  size_t o = 0;
  float* tlhs  = ws + o; o += (size_t)Bn * Tn * Vn;
  float* trhs  = ws + o; o += (size_t)Bn * Vn * Tn;
  float* prodT = ws + o; o += (size_t)Bn * Tn * Tn;
  float* TAtt  = ws + o; o += (size_t)Bn * Tn * Tn;
  float* xT    = ws + o; o += (size_t)Bn * Tn * Vn * Fn;
  float* slhs  = ws + o; o += (size_t)Bn * Vn * Tn;
  float* srhs  = ws + o; o += (size_t)Bn * Tn * Vn;
  float* prodS = ws + o; o += (size_t)Bn * Vn * Vn;
  float* SAtt  = ws + o; o += (size_t)Bn * Vn * Vn;
  _Float16* BfGL  = (_Float16*)(ws + o); o += (6 * 4 * 32 * 16) / 2;
  _Float16* BfSD  = (_Float16*)(ws + o); o += (6 * 4 * 32 * 16) / 2;
  _Float16* outGL = (_Float16*)(ws + o); o += ((size_t)Bn * Tn * Vn * FCn) / 2;
  _Float16* outSD = (_Float16*)(ws + o); o += ((size_t)Bn * Tn * Vn * FCn) / 2;

  const int BTV = Bn * Tn * Vn;

  // temporal attention
  k_tlhs<<<Bn * Tn, 64, 0, stream>>>(x, U1, U2, tlhs);
  k_trhs<<<(BTV + 255) / 256, 256, 0, stream>>>(x, U3, trhs);
  k_prodT<<<Bn * Tn, 128, 0, stream>>>(tlhs, trhs, prodT);
  k_sigT<<<Bn * Tn, 128, 0, stream>>>(prodT, V_e, b_e, TAtt);
  k_softcol<<<Bn * Tn, 128, 0, stream>>>(TAtt, Tn, Tn);
  k_xTAtt<<<Bn * Vn * Fn, 128, 0, stream>>>(x, TAtt, xT);

  // spatial attention
  k_slhs<<<Bn * Vn, 64, 0, stream>>>(xT, W1, W2, slhs);
  k_srhs<<<(BTV + 255) / 256, 256, 0, stream>>>(xT, W3, srhs);
  k_prodS<<<Bn * Vn, 64, 0, stream>>>(slhs, srhs, prodS);
  k_sigS<<<Bn * Vn, 64, 0, stream>>>(prodS, V_s, b_s, SAtt);
  k_softcol<<<Bn * Vn, 128, 0, stream>>>(SAtt, Vn, Vn);

  // graph learning + Chebyshev convs (f16 outputs)
  k_cheb<<<Bn * Tn, 256, 0, stream>>>(x, a_gl, SAtt, cheb, ThGL, ThSD, outGL, outSD);

  // WMMA temporal conv + LayerNorm
  k_prepW<<<1, 256, 0, stream>>>(WcGL, WcSD, BfGL, BfSD);
  float* yGL = (float*)d_out;
  float* ySD = yGL + (size_t)Bn * FTn * Tn * Vn;
  k_convln<<<BTV / 16, 32, 0, stream>>>(outGL, BfGL, bcGL, gGL, beGL, yGL);
  k_convln<<<BTV / 16, 32, 0, stream>>>(outSD, BfSD, bcSD, gSD, beSD, ySD);
}
